// BOTE_40939628266146
// MI455X (gfx1250) — matrix operand order repaired and checked by reading, not compiled, not run on hardware
//
#include <hip/hip_runtime.h>

// ---- sizes ----
#define BB   64
#define SS   512
#define LW   256
#define DD   768
#define RW   16384        // B*L rows
#define REDUC 400
#define RED_P 416         // padded to 13*32
#define HID  100
#define HID_P 128         // padded to 4*32 (col 100 = ones for biaffine A)
#define POL  4
#define TAGS 5
#define PH   400          // POL*HID
#define PH_P 416          // padded to 13*32

typedef __attribute__((ext_vector_type(16))) _Float16 v16h;
typedef __attribute__((ext_vector_type(8)))  _Float16 v8h;
typedef __attribute__((ext_vector_type(8)))  float    v8f;

// ---------------------------------------------------------------------------
// Operand load: 16 halves from row pointer p at element offset `off`,
// second 8-half group at +gap (A and B both use gap=8 at off 16*half: contiguous k = 16*half + s).
// ---------------------------------------------------------------------------
__device__ __forceinline__ v16h ld_op16(const _Float16* __restrict__ p, int off, int gap) {
  v8h lo = *(const v8h*)(p + off);
  v8h hi = *(const v8h*)(p + off + gap);
  return __builtin_shufflevector(lo, hi, 0,1,2,3,4,5,6,7,8,9,10,11,12,13,14,15);
}

__device__ __forceinline__ v8f wmma_f16(v16h a, v16h b, v8f c) {
  return __builtin_amdgcn_wmma_f32_16x16x32_f16(false, a, false, b, (short)0, c,
                                                false, false);
}

// 2x2 register-blocked accumulation: one wave owns a 32x32 C macro-tile.
struct Acc4 { v8f c00, c01, c10, c11; };

__device__ __forceinline__ Acc4 wmma_accum_2x2(const _Float16* __restrict__ A0,
                                               const _Float16* __restrict__ A1,
                                               const _Float16* __restrict__ B0,
                                               const _Float16* __restrict__ B1,
                                               int ksteps, int ka, int kb) {
  Acc4 c = {};
  for (int s = 0; s < ksteps; ++s) {
    int o = s * 32;
    v16h a0 = ld_op16(A0, o + kb, 8);
    v16h a1 = ld_op16(A1, o + kb, 8);
    v16h b0 = ld_op16(B0, o + kb, 8);
    v16h b1 = ld_op16(B1, o + kb, 8);
    c.c00 = wmma_f16(a0, b0, c.c00);
    c.c01 = wmma_f16(a0, b1, c.c01);
    c.c10 = wmma_f16(a1, b0, c.c10);
    c.c11 = wmma_f16(a1, b1, c.c11);
  }
  return c;
}

// ---------------------------------------------------------------------------
// Segment-mean pooling: bert [B,S,D] f32 + spans -> pooled f16 [RW, 768]
// ---------------------------------------------------------------------------
__global__ void pool_kernel(const float* __restrict__ bert,
                            const int* __restrict__ pos,
                            _Float16* __restrict__ out) {
  int w = blockIdx.x;                 // word row 0..16383 (= b*256 + l)
  int b = w >> 8;
  int s = pos[2 * w + 0];
  int e = pos[2 * w + 1];
  float inv = 1.0f / (float)(e - s + 1);
  const float* base = bert + ((size_t)b * SS + s) * DD;
  for (int c = threadIdx.x; c < DD; c += blockDim.x) {
    float acc = base[c];
    for (int r = 1; r <= e - s; ++r) acc += base[(size_t)r * DD + c];
    out[(size_t)w * DD + c] = (_Float16)(acc * inv);
  }
}

// ---------------------------------------------------------------------------
// Transpose + f32->f16 convert with zero padding: in [K,N] -> out [Npad,Kpad]
// ---------------------------------------------------------------------------
__global__ void wtrans_kernel(const float* __restrict__ in, int K, int N,
                              _Float16* __restrict__ out, int Kpad, int Npad) {
  int idx = blockIdx.x * blockDim.x + threadIdx.x;
  if (idx >= Npad * Kpad) return;
  int n = idx / Kpad, k = idx % Kpad;
  float v = (n < N && k < K) ? in[(size_t)k * N + n] : 0.0f;
  out[idx] = (_Float16)v;
}

// ---------------------------------------------------------------------------
// Epilogue helper: one 16x16 sub-tile, f16 store with bias+relu, n-guarded.
// ---------------------------------------------------------------------------
__device__ __forceinline__ void st_bias_relu(const v8f& acc, int mbase, int n,
                                             int Nreal, int half,
                                             const float* __restrict__ bias,
                                             _Float16* __restrict__ C, int ldc) {
  if (n < Nreal) {
    float bb = bias[n];
#pragma unroll
    for (int r = 0; r < 8; ++r) {
      float v = acc[r] + bb;
      v = v < 0.0f ? 0.0f : v;
      C[(size_t)(mbase + r + 8 * half) * ldc + n] = (_Float16)v;
    }
  }
}

// ---------------------------------------------------------------------------
// Generic WMMA GEMM (2x2 blocked): C_f16 = relu(A @ BT^T + bias)
// ---------------------------------------------------------------------------
__global__ void gemm_bias_relu_f16(const _Float16* __restrict__ A, int lda,
                                   const _Float16* __restrict__ BT, int ldb,
                                   const float* __restrict__ bias,
                                   _Float16* __restrict__ C, int ldc,
                                   int Mt2, int Nt2, int Nreal, int ksteps) {
  int wave = blockIdx.x * (blockDim.x >> 5) + (threadIdx.x >> 5);
  if (wave >= Mt2 * Nt2) return;
  int lane = threadIdx.x & 31;
  int half = lane >> 4, l16 = lane & 15;
  int tm = wave / Nt2, tn = wave % Nt2;
  const _Float16* A0 = A + (size_t)(tm * 32 + l16) * lda;
  const _Float16* A1 = A0 + (size_t)16 * lda;
  const _Float16* B0 = BT + (size_t)(tn * 32 + l16) * ldb;
  const _Float16* B1 = B0 + (size_t)16 * ldb;
  Acc4 c = wmma_accum_2x2(A0, A1, B0, B1, ksteps, half * 8, half * 16);
  int n0 = tn * 32 + l16, n1 = n0 + 16;
  int m0 = tm * 32, m1 = m0 + 16;
  st_bias_relu(c.c00, m0, n0, Nreal, half, bias, C, ldc);
  st_bias_relu(c.c01, m0, n1, Nreal, half, bias, C, ldc);
  st_bias_relu(c.c10, m1, n0, Nreal, half, bias, C, ldc);
  st_bias_relu(c.c11, m1, n1, Nreal, half, bias, C, ldc);
}

// ---------------------------------------------------------------------------
// Biaffine affine GEMM (2x2): a1[RW,128] @ W_bi^T[416,128] -> stored remapped:
// aff[(row*4 + p)*128 + h] f16  (p = n/100, h = n%100); n >= 400 dropped.
// ---------------------------------------------------------------------------
__device__ __forceinline__ void st_affine(const v8f& acc, int mbase, int n,
                                          int half, _Float16* __restrict__ aff) {
  if (n < PH) {
    int p = n / HID, hh = n % HID;
#pragma unroll
    for (int r = 0; r < 8; ++r) {
      int m = mbase + r + 8 * half;            // word row (b*256 + i)
      aff[((size_t)m * POL + p) * HID_P + hh] = (_Float16)acc[r];
    }
  }
}

__global__ void gemm_affine(const _Float16* __restrict__ A,
                            const _Float16* __restrict__ BT,
                            _Float16* __restrict__ aff,
                            int Mt2, int Nt2, int ksteps) {
  int wave = blockIdx.x * (blockDim.x >> 5) + (threadIdx.x >> 5);
  if (wave >= Mt2 * Nt2) return;
  int lane = threadIdx.x & 31;
  int half = lane >> 4, l16 = lane & 15;
  int tm = wave / Nt2, tn = wave % Nt2;
  const _Float16* A0 = A + (size_t)(tm * 32 + l16) * HID_P;
  const _Float16* A1 = A0 + (size_t)16 * HID_P;
  const _Float16* B0 = BT + (size_t)(tn * 32 + l16) * HID_P;
  const _Float16* B1 = B0 + (size_t)16 * HID_P;
  Acc4 c = wmma_accum_2x2(A0, A1, B0, B1, ksteps, half * 8, half * 16);
  int n0 = tn * 32 + l16, n1 = n0 + 16;
  int m0 = tm * 32, m1 = m0 + 16;
  st_affine(c.c00, m0, n0, half, aff);
  st_affine(c.c01, m0, n1, half, aff);
  st_affine(c.c10, m1, n0, half, aff);
  st_affine(c.c11, m1, n1, half, aff);
}

// ---------------------------------------------------------------------------
// Triplet batched GEMM (2x2): per batch b, C[j, i*4+p] = op[b,j,:].aff[b,i4p,:]
// f32 stores straight into triplet section of d_out (layout matches einsum).
// ---------------------------------------------------------------------------
__device__ __forceinline__ void st_f32(const v8f& acc, int mbase, int n, int half,
                                       float* __restrict__ outB, int ldo) {
#pragma unroll
  for (int r = 0; r < 8; ++r) {
    int m = mbase + r + 8 * half;
    outB[(size_t)m * ldo + n] = acc[r];
  }
}

__global__ void gemm_triplet(const _Float16* __restrict__ opAct,
                             const _Float16* __restrict__ aff,
                             float* __restrict__ out) {
  int b = blockIdx.z;
  int wave = blockIdx.x * (blockDim.x >> 5) + (threadIdx.x >> 5);
  const int Nt2 = (LW * POL) / 32;       // 32
  const int Mt2 = LW / 32;               // 8
  if (wave >= Mt2 * Nt2) return;
  int lane = threadIdx.x & 31;
  int half = lane >> 4, l16 = lane & 15;
  int tm = wave / Nt2, tn = wave % Nt2;
  const _Float16* A0 = opAct + ((size_t)b * LW + tm * 32 + l16) * HID_P;
  const _Float16* A1 = A0 + (size_t)16 * HID_P;
  const _Float16* B0 = aff + ((size_t)b * LW * POL + tn * 32 + l16) * HID_P;
  const _Float16* B1 = B0 + (size_t)16 * HID_P;
  Acc4 c = wmma_accum_2x2(A0, A1, B0, B1, HID_P / 32, half * 8, half * 16);
  float* outB = out + (size_t)b * LW * LW * POL;
  int n0 = tn * 32 + l16, n1 = n0 + 16;
  int m0 = tm * 32, m1 = m0 + 16;
  st_f32(c.c00, m0, n0, half, outB, LW * POL);
  st_f32(c.c01, m0, n1, half, outB, LW * POL);
  st_f32(c.c10, m1, n0, half, outB, LW * POL);
  st_f32(c.c11, m1, n1, half, outB, LW * POL);
}

// ---------------------------------------------------------------------------
// Set biaffine ones column: ap[row*128 + 100] = 1.0
// ---------------------------------------------------------------------------
__global__ void set_ones_kernel(_Float16* __restrict__ ap) {
  int r = blockIdx.x * blockDim.x + threadIdx.x;
  if (r < RW) ap[(size_t)r * HID_P + HID] = (_Float16)1.0f;
}

// ---------------------------------------------------------------------------
// Tag heads: out[row,t] = bias[t] + sum_h act[row,h] * W[h,t]   (100 -> 5)
// ---------------------------------------------------------------------------
__global__ void tag_kernel(const _Float16* __restrict__ act,
                           const float* __restrict__ W,
                           const float* __restrict__ bvec,
                           float* __restrict__ out) {
  int idx = blockIdx.x * blockDim.x + threadIdx.x;
  if (idx >= RW * TAGS) return;
  int row = idx / TAGS, t = idx % TAGS;
  const _Float16* a = act + (size_t)row * HID_P;
  float acc = bvec[t];
  for (int h = 0; h < HID; ++h) acc += (float)a[h] * W[h * TAGS + t];
  out[idx] = acc;
}

// ---------------------------------------------------------------------------
extern "C" void kernel_launch(void* const* d_in, const int* in_sizes, int n_in,
                              void* d_out, int out_size, void* d_ws, size_t ws_size,
                              hipStream_t stream) {
  const float* bert     = (const float*)d_in[0];
  const int*   pos      = (const int*)d_in[1];
  const float* W_reduc  = (const float*)d_in[2];
  const float* b_reduc  = (const float*)d_in[3];
  const float* W_ap     = (const float*)d_in[4];
  const float* b_ap     = (const float*)d_in[5];
  const float* W_op     = (const float*)d_in[6];
  const float* b_op     = (const float*)d_in[7];
  const float* W_ap_tag = (const float*)d_in[8];
  const float* b_ap_tag = (const float*)d_in[9];
  const float* W_op_tag = (const float*)d_in[10];
  const float* b_op_tag = (const float*)d_in[11];
  const float* W_bi     = (const float*)d_in[12];

  // workspace carve-out (256B aligned sub-buffers)
  char* ws = (char*)d_ws;
  size_t off = 0;
  auto carve = [&](size_t bytes) -> void* {
    void* p = ws + off;
    off = (off + bytes + 255) & ~(size_t)255;
    return p;
  };
  _Float16* pooled  = (_Float16*)carve((size_t)RW * DD * 2);      // [16384,768]
  _Float16* WrT     = (_Float16*)carve((size_t)RED_P * DD * 2);   // [416,768]
  _Float16* hbuf    = (_Float16*)carve((size_t)RW * RED_P * 2);   // [16384,416]
  _Float16* WapT    = (_Float16*)carve((size_t)HID_P * RED_P * 2);// [128,416]
  _Float16* WopT    = (_Float16*)carve((size_t)HID_P * RED_P * 2);// [128,416]
  _Float16* apAct   = (_Float16*)carve((size_t)RW * HID_P * 2);   // [16384,128]
  _Float16* opAct   = (_Float16*)carve((size_t)RW * HID_P * 2);   // [16384,128]
  _Float16* WbiT    = (_Float16*)carve((size_t)PH_P * HID_P * 2); // [416,128]
  _Float16* affBuf  = (_Float16*)carve((size_t)RW * POL * HID_P * 2); // [65536,128]
  size_t ws_used = off;

  float* out_ap      = (float*)d_out;                    // [16384,5]
  float* out_op      = out_ap + (size_t)RW * TAGS;       // [16384,5]
  float* out_triplet = out_op + (size_t)RW * TAGS;       // [64,256,256,4]

  // 0) zero workspace (guarantees all pad regions are zero)
  hipMemsetAsync(d_ws, 0, ws_used, stream);

  // 1) weight transposes (f32 -> f16, zero-padded to 32-multiples)
  {
    int n = RED_P * DD;
    wtrans_kernel<<<(n + 255) / 256, 256, 0, stream>>>(W_reduc, DD, REDUC, WrT, DD, RED_P);
  }
  {
    int n = HID_P * RED_P;
    wtrans_kernel<<<(n + 255) / 256, 256, 0, stream>>>(W_ap, REDUC, HID, WapT, RED_P, HID_P);
    wtrans_kernel<<<(n + 255) / 256, 256, 0, stream>>>(W_op, REDUC, HID, WopT, RED_P, HID_P);
  }
  {
    int n = PH_P * HID_P;
    wtrans_kernel<<<(n + 255) / 256, 256, 0, stream>>>(W_bi, HID + 1, PH, WbiT, HID_P, PH_P);
  }

  // 2) segment-mean pooling -> f16
  pool_kernel<<<RW, 128, 0, stream>>>(bert, pos, pooled);

  // 3) GEMM1: h = relu(pooled @ W_reduc + b)   [16384,768]x[768,400]
  {
    int Mt2 = RW / 32, Nt2 = RED_P / 32;              // 512 x 13
    int waves = Mt2 * Nt2;
    gemm_bias_relu_f16<<<(waves + 7) / 8, 256, 0, stream>>>(
        pooled, DD, WrT, DD, b_reduc, hbuf, RED_P, Mt2, Nt2, REDUC, DD / 32);
  }

  // 4) GEMM2/3: ap / op = relu(h @ W)  [16384,416]x[416,128(->100)]
  {
    int Mt2 = RW / 32, Nt2 = HID_P / 32;              // 512 x 4
    int waves = Mt2 * Nt2;
    gemm_bias_relu_f16<<<(waves + 7) / 8, 256, 0, stream>>>(
        hbuf, RED_P, WapT, RED_P, b_ap, apAct, HID_P, Mt2, Nt2, HID, RED_P / 32);
    gemm_bias_relu_f16<<<(waves + 7) / 8, 256, 0, stream>>>(
        hbuf, RED_P, WopT, RED_P, b_op, opAct, HID_P, Mt2, Nt2, HID, RED_P / 32);
  }

  // 5) biaffine ones column
  set_ones_kernel<<<(RW + 255) / 256, 256, 0, stream>>>(apAct);

  // 6) GEMM4: affine = a1 @ W_bi   [16384,128]x[128,416(->400)] -> remapped f16
  {
    int Mt2 = RW / 32, Nt2 = PH_P / 32;               // 512 x 13
    int waves = Mt2 * Nt2;
    gemm_affine<<<(waves + 7) / 8, 256, 0, stream>>>(
        apAct, WbiT, affBuf, Mt2, Nt2, HID_P / 32);
  }

  // 7) GEMM5: triplet[b,j,i,p] = op[b,j,:] . affine[b,i,p,:]  -> f32 d_out
  {
    int wavesPerBatch = (LW / 32) * ((LW * POL) / 32); // 8*32 = 256
    dim3 grid((wavesPerBatch + 7) / 8, 1, BB);
    gemm_triplet<<<grid, 256, 0, stream>>>(opAct, affBuf, out_triplet);
  }

  // 8) tag heads
  {
    int n = RW * TAGS;
    tag_kernel<<<(n + 255) / 256, 256, 0, stream>>>(apAct, W_ap_tag, b_ap_tag, out_ap);
    tag_kernel<<<(n + 255) / 256, 256, 0, stream>>>(opAct, W_op_tag, b_op_tag, out_op);
  }
}
